// GCN_83657372991743
// MI455X (gfx1250) — matrix operand order repaired and checked by reading, not compiled, not run on hardware
//
#include <hip/hip_runtime.h>
#include <hip/hip_bf16.h>

// ---------------------------------------------------------------------------
// Types for WMMA (gfx1250, wave32): V_WMMA_F32_16X16X32_BF16
// ---------------------------------------------------------------------------
typedef __attribute__((ext_vector_type(16))) __bf16        bf16x16;
typedef __attribute__((ext_vector_type(8)))  float         f32x8;
typedef __attribute__((ext_vector_type(8)))  unsigned int  u32x8;

// Pack two f32 -> {bf16(hi),bf16(lo)} with round-to-nearest-even.
// Final merge is a single v_perm_b32 (bytes 3:2 of each rounded word).
__device__ __forceinline__ unsigned pack_bf16x2(float lo, float hi) {
    unsigned ul = __float_as_uint(lo);
    unsigned uh = __float_as_uint(hi);
    ul += 0x7FFFu + ((ul >> 16) & 1u);
    uh += 0x7FFFu + ((uh >> 16) & 1u);
    return __builtin_amdgcn_perm(uh, ul, 0x07060302u);  // {uh[31:16], ul[31:16]}
}

// ---------------------------------------------------------------------------
// Dense GEMM: C[M,Nout] = (RELU ? relu(A) : A)[M,K] @ B[K,Nout]
// f32 in/out, bf16 WMMA compute with f32 accumulation.
// Block tile 128x64, BK=32. 256 threads = 8 waves; wave grid 4(m) x 2(n),
// each wave owns a 32x32 patch = 2x2 WMMA 16x16 tiles.
// Software-pipelined: double-buffered LDS, one barrier per K-step; global
// loads for step i+1 issue before the WMMAs of step i.
// ---------------------------------------------------------------------------
#define BM 128
#define BN 64
#define BK 32

template <bool RELU>
__global__ __launch_bounds__(256)
void gcn_gemm_wmma_bf16(const float* __restrict__ A, const float* __restrict__ B,
                        float* __restrict__ C,
                        int M, int K, int Nout, int lda, int ldb, int ldc)
{
    // LDS tiles stored as packed bf16 pairs {k, k+1} in one u32; double buffered
    __shared__ unsigned int Asl[2][BM][BK / 2];  // 2 * 8 KB
    __shared__ unsigned int Bsl[2][BN][BK / 2];  // 2 * 4 KB

    const int t    = threadIdx.x;
    const int lane = t & 31;
    const int wave = t >> 5;
    const int wm   = wave & 3;    // 0..3  -> m offset wm*32
    const int wn   = wave >> 2;   // 0..1  -> n offset wn*32
    const int lh   = lane >> 4;   // lane-half select (0: lanes 0-15, 1: 16-31)
    const int l16  = lane & 15;

    const int block_m = blockIdx.x * BM;
    const int block_n = blockIdx.y * BN;
    const bool mfull  = (block_m + BM <= M);   // block-uniform fast path

    f32x8 acc[2][2] = {};

    float4 ar[4];   // staged A rows (4 floats each)
    float2 br[4];   // staged B k-pairs

    // ---- stage global -> registers for tile starting at k0 ----
    auto load_tiles = [&](int k0) {
#pragma unroll
        for (int p = 0; p < 4; ++p) {
            int idx = t + p * 256;      // 0..1023, 4 consecutive floats each
            int r   = idx >> 3;         // 0..127
            int c   = (idx & 7) << 2;   // 0,4,...,28
            int gr  = block_m + r;
            if (mfull || gr < M)
                ar[p] = *(const float4*)(A + (size_t)gr * lda + k0 + c);
            else
                ar[p] = make_float4(0.f, 0.f, 0.f, 0.f);
        }
#pragma unroll
        for (int p = 0; p < 4; ++p) {
            int j  = t + p * 256;       // 0..1023
            int kp = j >> 6;            // 0..15
            int n  = j & 63;
            br[p].x = B[(size_t)(k0 + 2 * kp + 0) * ldb + block_n + n];
            br[p].y = B[(size_t)(k0 + 2 * kp + 1) * ldb + block_n + n];
        }
    };
    // ---- convert + store registers -> LDS buffer `buf` ----
    auto store_tiles = [&](int buf) {
#pragma unroll
        for (int p = 0; p < 4; ++p) {
            int idx = t + p * 256;
            int r   = idx >> 3;
            int c   = (idx & 7) << 2;
            float4 v = ar[p];
            if (RELU) {
                v.x = fmaxf(v.x, 0.f); v.y = fmaxf(v.y, 0.f);
                v.z = fmaxf(v.z, 0.f); v.w = fmaxf(v.w, 0.f);
            }
            Asl[buf][r][(c >> 1) + 0] = pack_bf16x2(v.x, v.y);
            Asl[buf][r][(c >> 1) + 1] = pack_bf16x2(v.z, v.w);
        }
#pragma unroll
        for (int p = 0; p < 4; ++p) {
            int j  = t + p * 256;
            int kp = j >> 6;
            int n  = j & 63;
            Bsl[buf][n][kp] = pack_bf16x2(br[p].x, br[p].y);
        }
    };

    // prologue: tile 0 into LDS buffer 0
    load_tiles(0);
    store_tiles(0);

    const int niter = K / BK;
    for (int i = 0; i < niter; ++i) {
        __syncthreads();                       // LDS[cur] ready; LDS[next] free
        const int cur = i & 1;

        if (i + 1 < niter) load_tiles((i + 1) * BK);   // overlap with WMMAs

        // ---- fragments per the 16-bit WMMA VGPR layouts ----
        // A 16x32: lane = M%16; reg v K pair: 2*(v&3) + 8*laneHalf + 16*(v>=4)
        // B 32x16: lane = N%16; reg v K pair: 2*(v&3) + 8*laneHalf + 16*(v>=4)
        u32x8 afrag[2], bfrag[2];
#pragma unroll
        for (int tm = 0; tm < 2; ++tm) {
            int row = wm * 32 + tm * 16 + l16;
#pragma unroll
            for (int v = 0; v < 8; ++v) {
                int kp = (v & 3) + lh * 4 + (v >> 2) * 8;
                afrag[tm][v] = Asl[cur][row][kp];
            }
        }
#pragma unroll
        for (int tn = 0; tn < 2; ++tn) {
            int n = wn * 32 + tn * 16 + l16;
#pragma unroll
            for (int v = 0; v < 8; ++v) {
                int kp = (v & 3) + lh * 4 + (v >> 2) * 8;
                bfrag[tn][v] = Bsl[cur][n][kp];
            }
        }
#pragma unroll
        for (int tm = 0; tm < 2; ++tm)
#pragma unroll
            for (int tn = 0; tn < 2; ++tn)
                acc[tm][tn] = __builtin_amdgcn_wmma_f32_16x16x32_bf16(
                    false, __builtin_bit_cast(bf16x16, afrag[tm]),
                    false, __builtin_bit_cast(bf16x16, bfrag[tn]),
                    (short)0, acc[tm][tn], false, false);

        if (i + 1 < niter) store_tiles(cur ^ 1);
    }

    // ---- store C: VGPR r -> M = r + 8*laneHalf, N = lane%16 ----
#pragma unroll
    for (int tm = 0; tm < 2; ++tm) {
        int m_base = block_m + wm * 32 + tm * 16 + lh * 8;
#pragma unroll
        for (int tn = 0; tn < 2; ++tn) {
            int n = block_n + wn * 32 + tn * 16 + l16;
#pragma unroll
            for (int r = 0; r < 8; ++r) {
                int row = m_base + r;
                if (mfull || row < M) C[(size_t)row * ldc + n] = acc[tm][tn][r];
            }
        }
    }
}

// ---------------------------------------------------------------------------
// out[i*F + f] = bias[f]   (accumulator init; F is a power of two)
// ---------------------------------------------------------------------------
__global__ __launch_bounds__(256)
void gcn_init_bias(float* __restrict__ out, const float* __restrict__ bias,
                   int total, int F)
{
    int i = blockIdx.x * blockDim.x + threadIdx.x;
    if (i < total) out[i] = bias[i & (F - 1)];
}

// ---------------------------------------------------------------------------
// SpMM scatter: out[row[e], :] += val[e] * in[col[e], :]
// One wave32 per edge (grid-stride). Gathered rows live in L2 (102 MB < 192 MB),
// atomics resolve at L2 (global_atomic_add_f32). Edge stream prefetched ahead.
// ---------------------------------------------------------------------------
__global__ __launch_bounds__(256)
void gcn_spmm_atomic(const int* __restrict__ rows, const int* __restrict__ cols,
                     const float* __restrict__ vals,
                     const float* __restrict__ in, float* __restrict__ out,
                     int E, int F)
{
    int wave   = (blockIdx.x * blockDim.x + threadIdx.x) >> 5;
    int lane   = threadIdx.x & 31;
    int nwaves = (gridDim.x * blockDim.x) >> 5;
    for (int e = wave; e < E; e += nwaves) {
        if (e + nwaves < E) {                    // stream-ahead prefetch
            __builtin_prefetch(rows + e + nwaves, 0, 0);
            __builtin_prefetch(cols + e + nwaves, 0, 0);
            __builtin_prefetch(vals + e + nwaves, 0, 0);
        }
        int   r = rows[e];
        int   c = cols[e];
        float w = vals[e];
        const float* src = in  + (size_t)c * F;
        float*       dst = out + (size_t)r * F;
        for (int f = lane * 4; f < F; f += 128) {
            float4 v = *(const float4*)(src + f);
            atomicAdd(dst + f + 0, w * v.x);
            atomicAdd(dst + f + 1, w * v.y);
            atomicAdd(dst + f + 2, w * v.z);
            atomicAdd(dst + f + 3, w * v.w);
        }
    }
}

// ---------------------------------------------------------------------------
// Row-wise log_softmax over 64 columns; one wave32 per row, 2 values/lane.
// ---------------------------------------------------------------------------
__global__ __launch_bounds__(256)
void gcn_log_softmax64(const float* __restrict__ in, float* __restrict__ out, int Nrows)
{
    int wave = (blockIdx.x * blockDim.x + threadIdx.x) >> 5;
    int lane = threadIdx.x & 31;
    if (wave >= Nrows) return;
    const float* x = in + (size_t)wave * 64;
    float a = x[lane], b = x[lane + 32];
    float m = fmaxf(a, b);
#pragma unroll
    for (int off = 16; off; off >>= 1) m = fmaxf(m, __shfl_xor(m, off, 32));
    float s = __expf(a - m) + __expf(b - m);
#pragma unroll
    for (int off = 16; off; off >>= 1) s += __shfl_xor(s, off, 32);
    float l = m + __logf(s);
    float* o = out + (size_t)wave * 64;
    o[lane]      = a - l;
    o[lane + 32] = b - l;
}

// ---------------------------------------------------------------------------
// Orchestration: GEMM -> (bias-init + SpMM-atomics) per layer, ReLU fused into
// the next GEMM's A-load; final log-softmax. Two ping-pong scratch buffers.
// ---------------------------------------------------------------------------
extern "C" void kernel_launch(void* const* d_in, const int* in_sizes, int n_in,
                              void* d_out, int out_size, void* d_ws, size_t ws_size,
                              hipStream_t stream)
{
    const float* x    = (const float*)d_in[0];
    const int*   erow = (const int*)  d_in[1];
    const int*   ecol = (const int*)  d_in[2];
    const float* ev   = (const float*)d_in[3];
    const float* W1   = (const float*)d_in[4];
    const float* b1   = (const float*)d_in[5];
    const float* W2   = (const float*)d_in[6];
    const float* b2   = (const float*)d_in[7];
    const float* W3   = (const float*)d_in[8];
    const float* b3   = (const float*)d_in[9];
    (void)n_in; (void)out_size; (void)ws_size;

    const int N    = in_sizes[0] / 512;           // 100000
    const int E    = in_sizes[1];                 // 3200000
    const int Mpad = ((N + BM - 1) / BM) * BM;
    const size_t S = (size_t)Mpad * 256 * sizeof(float);

    float* bufC = (float*)d_ws;                   // GEMM outputs (support)
    float* bufH = (float*)((char*)d_ws + S);      // SpMM accumulators

    dim3 blk(256);
    const int spmm_blocks = 4096;

    // ---- Layer 1: C = x @ W1 ; H = b1 + A*C ----
    gcn_gemm_wmma_bf16<false><<<dim3(Mpad / BM, 256 / BN), blk, 0, stream>>>(
        x, W1, bufC, N, 512, 256, 512, 256, 256);
    gcn_init_bias<<<(N * 256 + 255) / 256, blk, 0, stream>>>(bufH, b1, N * 256, 256);
    gcn_spmm_atomic<<<spmm_blocks, blk, 0, stream>>>(erow, ecol, ev, bufC, bufH, E, 256);

    // ---- Layer 2: C = relu(H) @ W2 ; H = b2 + A*C ----
    gcn_gemm_wmma_bf16<true><<<dim3(Mpad / BM, 256 / BN), blk, 0, stream>>>(
        bufH, W2, bufC, N, 256, 256, 256, 256, 256);
    gcn_init_bias<<<(N * 256 + 255) / 256, blk, 0, stream>>>(bufH, b2, N * 256, 256);
    gcn_spmm_atomic<<<spmm_blocks, blk, 0, stream>>>(erow, ecol, ev, bufC, bufH, E, 256);

    // ---- Layer 3: C = relu(H) @ W3 ; H = b3 + A*C ----
    gcn_gemm_wmma_bf16<true><<<dim3(Mpad / BM, 64 / BN), blk, 0, stream>>>(
        bufH, W3, bufC, N, 256, 64, 256, 64, 64);
    gcn_init_bias<<<(N * 64 + 255) / 256, blk, 0, stream>>>(bufH, b3, N * 64, 64);
    gcn_spmm_atomic<<<spmm_blocks, blk, 0, stream>>>(erow, ecol, ev, bufC, bufH, E, 64);

    // ---- log_softmax ----
    gcn_log_softmax64<<<(N * 32 + 255) / 256, blk, 0, stream>>>(bufH, (float*)d_out, N);
}
